// TransitionMatrix_824633720864
// MI455X (gfx1250) — matrix operand order repaired and checked by reading, not compiled, not run on hardware
//
#include <hip/hip_runtime.h>
#include <hip/hip_bf16.h>
#include <math.h>

typedef __attribute__((ext_vector_type(16))) _Float16 v16h;
typedef __attribute__((ext_vector_type(8)))  float    v8f;
typedef __attribute__((ext_vector_type(4)))  float    v4f;

#define LOG2E 1.44269504088896340736f
#define LN2   0.69314718055994530942f

// ---------------------------------------------------------------------------
// Kernel 1: T[i,k] = softmax(matrix, axis=0) in f16, row-major (2 KB -> d_ws).
// One wave; thread k owns column k.
// ---------------------------------------------------------------------------
__global__ void build_T_kernel(const float* __restrict__ mat,
                               _Float16* __restrict__ Tf16) {
    int k = threadIdx.x;
    if (k >= 32) return;
    float col[32];
    float cmax = -INFINITY;
#pragma unroll
    for (int i = 0; i < 32; ++i) {
        col[i] = mat[i * 32 + k];
        cmax = fmaxf(cmax, col[i]);
    }
    float s = 0.0f;
#pragma unroll
    for (int i = 0; i < 32; ++i) {
        col[i] = __builtin_amdgcn_exp2f((col[i] - cmax) * LOG2E);
        s += col[i];
    }
    float inv = 1.0f / s;
#pragma unroll
    for (int i = 0; i < 32; ++i) {
        Tf16[i * 32 + k] = (_Float16)(col[i] * inv);
    }
}

// ---------------------------------------------------------------------------
// Kernel 2: streaming log-domain matmul.
// Each wave: 16 batch rows. out[b,i] = log( sum_k T[i,k]*exp(la[b,k]-m_b) ) + m_b
// via two v_wmma_f32_16x16x32_f16 (N=32 -> two 16-col tiles, K=32 in one shot).
// ---------------------------------------------------------------------------
__global__ void __launch_bounds__(256)
log_matmul_kernel(const float* __restrict__ la,
                  const _Float16* __restrict__ Tf16,
                  float* __restrict__ out, int Bn) {
    const int lane = threadIdx.x & 31;
    const int wave = blockIdx.x * 8 + (threadIdx.x >> 5);
    const int b0   = wave * 16;
    const int h    = lane >> 4;   // half-wave select (K-group in A/B layouts)
    const int m16  = lane & 15;   // A-row / B-col index

    if (b0 + 16 > Bn) return;     // wave-uniform exit (B is a multiple of 16)

    // ---- Load A half-rows in contiguous k order (k = 16h + s), matching B ----
    // lane (h,m16) holds K = {16h..16h+15}
    const float* rowp = la + (size_t)(b0 + m16) * 32;
    float v[16];
    *(v4f*)&v[0]  = *(const v4f*)(rowp + 16 * h);
    *(v4f*)&v[4]  = *(const v4f*)(rowp + 16 * h + 4);
    *(v4f*)&v[8]  = *(const v4f*)(rowp + 16 * h + 8);
    *(v4f*)&v[12] = *(const v4f*)(rowp + 16 * h + 12);

    // ---- Per-row max: lane and lane^16 together hold the full row ----
    float lmax = v[0];
#pragma unroll
    for (int j = 1; j < 16; ++j) lmax = fmaxf(lmax, v[j]);
    float mrow = fmaxf(lmax, __shfl_xor(lmax, 16, 32));

    // ---- A fragment: exp(v - m) = exp2(fma(v, log2e, -m*log2e)) in f16 ----
    const float nm2 = -mrow * LOG2E;
    v16h a;
#pragma unroll
    for (int j = 0; j < 16; ++j)
        a[j] = (_Float16)__builtin_amdgcn_exp2f(__builtin_fmaf(v[j], LOG2E, nm2));

    // ---- B fragments: B[k,n] = T[n,k]; lane n=m16 reads 16 contiguous
    //      halves of T row n starting at K = 16h (32 B, coalesced) ----
    v16h bt0 = *(const v16h*)(Tf16 + (size_t)m16 * 32 + 16 * h);
    v16h bt1 = *(const v16h*)(Tf16 + (size_t)(m16 + 16) * 32 + 16 * h);

    v8f zero = {};
    v8f d0 = __builtin_amdgcn_wmma_f32_16x16x32_f16(
        false, a, false, bt0, (short)0, zero, false, false);
    v8f d1 = __builtin_amdgcn_wmma_f32_16x16x32_f16(
        false, a, false, bt1, (short)0, zero, false, false);

    // ---- Epilogue: out = log2(C)*ln2 + m ; C row M = r + 8h, col N = m16 ----
#pragma unroll
    for (int r = 0; r < 8; ++r) {
        float mr = __shfl(mrow, r + 8 * h, 32);   // ds_bpermute broadcast
        size_t brow = (size_t)(b0 + r + 8 * h) * 32;
        out[brow + m16] =
            __builtin_fmaf(__builtin_amdgcn_logf(d0[r]), LN2, mr);
        out[brow + 16 + m16] =
            __builtin_fmaf(__builtin_amdgcn_logf(d1[r]), LN2, mr);
    }
}

// ---------------------------------------------------------------------------
extern "C" void kernel_launch(void* const* d_in, const int* in_sizes, int n_in,
                              void* d_out, int out_size, void* d_ws, size_t ws_size,
                              hipStream_t stream) {
    const float* matrix = (const float*)d_in[0];   // (32,32) f32
    const float* la     = (const float*)d_in[1];   // (B,32) f32
    float* out          = (float*)d_out;           // (B,32) f32
    _Float16* Tf16      = (_Float16*)d_ws;         // 32*32 f16 = 2 KB scratch

    int Bn = in_sizes[1] / 32;

    build_T_kernel<<<1, 32, 0, stream>>>(matrix, Tf16);

    int tiles  = Bn / 16;               // one 16-row tile per wave
    int blocks = (tiles + 7) / 8;       // 8 waves per 256-thread block
    log_matmul_kernel<<<blocks, 256, 0, stream>>>(la, Tf16, out, Bn);
}
